// Transformer_9036611191235
// MI455X (gfx1250) — matrix operand order repaired and checked by reading, not compiled, not run on hardware
//
#include <hip/hip_runtime.h>

// ---------------------------------------------------------------------------
// MI455X (gfx1250, wave32) transformer encoder.
// GEMMs: C[M,N] = A[M,K](f16) * B[N,K](f16)^T via v_wmma_f32_16x16x32_f16.
// Tile staging uses GLOBAL_LOAD_ASYNC_TO_LDS_B128 (ASYNCcnt-tracked) with
// double-buffered LDS so the async copy engine runs ahead of the WMMA pipe.
// Attention (QK^T -> softmax -> AV) fused per (b,n,head): scores never hit HBM.
// ---------------------------------------------------------------------------

typedef _Float16 v16h __attribute__((ext_vector_type(16)));
typedef _Float16 h8   __attribute__((ext_vector_type(8)));
typedef float    v8f  __attribute__((ext_vector_type(8)));

#define Bb   8
#define Nn   32
#define Ll   128
#define Dd   256
#define Hh   8
#define Ee   32
#define DFF  1024
#define NTOK 32768          // B*N*L
#define NLAY 2

__device__ inline v16h mk16(h8 lo, h8 hi) {
  return __builtin_shufflevector(lo, hi, 0,1,2,3,4,5,6,7,8,9,10,11,12,13,14,15);
}
__device__ inline v8f wmma16(v16h a, v16h b, v8f c) {
  return __builtin_amdgcn_wmma_f32_16x16x32_f16(false, a, false, b, (short)0, c, false, false);
}

// --- CDNA5 async global->LDS copy (ASYNCcnt-tracked, bypasses VGPRs) --------
// LDS operand is the low 32 bits of the flat address (LDS aperture: addr[31:0]).
__device__ __forceinline__ unsigned lds_addr32(const void* p) {
  return (unsigned)(unsigned long long)p;
}
__device__ __forceinline__ void async_b128(unsigned lds, const void* g) {
  asm volatile("global_load_async_to_lds_b128 %0, %1, off"
               :: "v"(lds), "v"((unsigned long long)g) : "memory");
}
__device__ __forceinline__ void wait_async0() {
  asm volatile("s_wait_asynccnt 0" ::: "memory");
}
__device__ __forceinline__ void wait_async4() {   // allow next buffer's 4 in flight
  asm volatile("s_wait_asynccnt 4" ::: "memory");
}

// ---------------- one-time weight prep -------------------------------------

// Wq/Wk/Wv [NLAY][Nn][d][o] (f32) -> Wqkv_t [NLAY*Nn][768][256] f16, rows=o(+sect*256), cols=d
__global__ __launch_bounds__(256) void prep_qkvw(const float* __restrict__ src,
                                                 _Float16* __restrict__ dst, int sect) {
  long idx = (long)blockIdx.x * 256 + threadIdx.x;
  if (idx >= (long)NLAY * Nn * Dd * Dd) return;
  long ln = idx >> 16;                                  // layer*32+node
  int e  = (int)(idx & 65535);
  int d  = e >> 8, o = e & 255;
  dst[(ln * 768 + sect * 256 + o) * 256 + d] = (_Float16)src[idx];
}

__global__ __launch_bounds__(256) void prep_cvt(const float* __restrict__ src,
                                                _Float16* __restrict__ dst, long n) {
  long i = (long)blockIdx.x * 256 + threadIdx.x;
  if (i < n) dst[i] = (_Float16)src[i];
}

// bq/bk/bv [NLAY][Nn][256] -> bqkv [NLAY*Nn][768]
__global__ __launch_bounds__(256) void prep_bias(const float* __restrict__ src,
                                                 float* __restrict__ dst, int sect) {
  int idx = blockIdx.x * 256 + threadIdx.x;
  if (idx >= NLAY * Nn * 256) return;
  int ln = idx >> 8, o = idx & 255;
  dst[ln * 768 + sect * 256 + o] = src[idx];
}

// ---------------- embedding -------------------------------------------------
__global__ __launch_bounds__(256) void embed_kernel(const float* __restrict__ x,
                                                    const float* __restrict__ W,
                                                    const float* __restrict__ be,
                                                    const float* __restrict__ pos,
                                                    float* __restrict__ h,
                                                    _Float16* __restrict__ hf) {
  __shared__ float sx[16];
  long tkn = blockIdx.x;
  int c = threadIdx.x;
  if (c < 16) sx[c] = x[tkn * 16 + c];
  __syncthreads();
  float acc = be[c] + pos[(tkn & (Ll - 1)) * Dd + c];
#pragma unroll
  for (int i = 0; i < 16; ++i) acc += sx[i] * W[i * Dd + c];
  h[tkn * Dd + c]  = acc;
  hf[tkn * Dd + c] = (_Float16)acc;
}

// ---------------- generic WMMA GEMM (A[M,K] * B[N,K]^T) ---------------------
// 128x128 block tile, 8 waves of 32x64, K-step 32, double-buffered async LDS.
enum { EPI_F32 = 0, EPI_RELU_F16 = 1, EPI_QKV = 2 };

template <int EPI>
__global__ __launch_bounds__(256) void gemm_nt(const _Float16* __restrict__ A,
                                               const _Float16* __restrict__ B,
                                               const float* __restrict__ bias,
                                               float* __restrict__ outF,
                                               _Float16* __restrict__ o16a, // relu dst or q16
                                               _Float16* __restrict__ o16b, // k16
                                               _Float16* __restrict__ o16c, // v_t
                                               int M, int N, int K) {
  __shared__ _Float16 As[2][128 * 40];
  __shared__ _Float16 Bs[2][128 * 40];
  const int tid = threadIdx.x, lane = tid & 31, wave = tid >> 5;
  const int wm = wave >> 1, wn = wave & 1;            // 4 x 2 wave grid
  const int bm = blockIdx.x, bn = blockIdx.y;
  const int node = (EPI == EPI_QKV) ? (int)blockIdx.z : 0;

  const int lr = tid >> 1, khalf = tid & 1;           // cooperative tile load
  long arow;
  {
    int r = bm * 128 + lr;
    if (EPI == EPI_QKV) arow = (long)(r >> 7) * (Nn * Ll) + (long)node * Ll + (r & 127);
    else                arow = r;
  }
  const _Float16* Aptr = A + arow * (long)K + khalf * 16;
  const _Float16* Bptr = B + ((long)node * 768 + bn * 128 + lr) * (long)K + khalf * 16;
  const unsigned aLds = lds_addr32(&As[0][lr * 40 + khalf * 16]);
  const unsigned bLds = lds_addr32(&Bs[0][lr * 40 + khalf * 16]);
  const unsigned bufStride = 128 * 40 * 2;            // bytes between buffers

  // async prefetch of first two k-tiles (4 b128 per thread per tile)
#pragma unroll
  for (int p = 0; p < 2; ++p) {
    async_b128(aLds + p * bufStride,      Aptr + p * 32);
    async_b128(aLds + p * bufStride + 16, Aptr + p * 32 + 8);
    async_b128(bLds + p * bufStride,      Bptr + p * 32);
    async_b128(bLds + p * bufStride + 16, Bptr + p * 32 + 8);
  }

  v8f acc[2][4];
#pragma unroll
  for (int mt = 0; mt < 2; ++mt)
#pragma unroll
    for (int nt = 0; nt < 4; ++nt)
#pragma unroll
      for (int i = 0; i < 8; ++i) acc[mt][nt][i] = 0.f;

  const int rowl = lane & 15;
  const int ka0  = (lane < 16) ? 0 : 8;               // A-fragment K chunk base
  const int kb0  = (lane < 16) ? 0 : 8;               // B-fragment K base

  const int niter = K >> 5;
  for (int it = 0; it < niter; ++it) {
    const int buf = it & 1;
    // async loads complete in order: <=4 outstanding means this buffer landed
    if (it == niter - 1) wait_async0(); else wait_async4();
    __syncthreads();

    v16h af[2], bf[4];
#pragma unroll
    for (int mt = 0; mt < 2; ++mt) {
      const _Float16* p = &As[buf][(wm * 32 + mt * 16 + rowl) * 40 + ka0];
      af[mt] = mk16(*(const h8*)p, *(const h8*)(p + 16));
    }
#pragma unroll
    for (int nt = 0; nt < 4; ++nt) {
      const _Float16* p = &Bs[buf][(wn * 64 + nt * 16 + rowl) * 40 + kb0];
      bf[nt] = mk16(*(const h8*)p, *(const h8*)(p + 16));
    }
#pragma unroll
    for (int mt = 0; mt < 2; ++mt)
#pragma unroll
      for (int nt = 0; nt < 4; ++nt) acc[mt][nt] = wmma16(af[mt], bf[nt], acc[mt][nt]);

    __syncthreads();                                   // all waves done with buf
    if (it + 2 < niter) {                              // prefetch iteration it+2
      const int kk = (it + 2) * 32;
      async_b128(aLds + buf * bufStride,      Aptr + kk);
      async_b128(aLds + buf * bufStride + 16, Aptr + kk + 8);
      async_b128(bLds + buf * bufStride,      Bptr + kk);
      async_b128(bLds + buf * bufStride + 16, Bptr + kk + 8);
    }
  }

  // epilogue (C layout: vgpr i -> row i + (lane>=16?8:0), col lane&15)
  const int rbase = bm * 128 + wm * 32;
  const int cbase = bn * 128 + wn * 64;
  const int rofs  = (lane >= 16) ? 8 : 0;
#pragma unroll
  for (int mt = 0; mt < 2; ++mt)
#pragma unroll
    for (int nt = 0; nt < 4; ++nt)
#pragma unroll
      for (int i = 0; i < 8; ++i) {
        int r = rbase + mt * 16 + i + rofs;
        int c = cbase + nt * 16 + (lane & 15);
        float v = acc[mt][nt][i];
        if (EPI == EPI_F32) {
          if (bias) v += bias[c];
          outF[(long)r * N + c] = v;
        } else if (EPI == EPI_RELU_F16) {
          o16a[(long)r * N + c] = (_Float16)fmaxf(v, 0.f);
        } else { // EPI_QKV: r is node-local row, N==768
          v += bias[(long)node * 768 + c];
          long token = (long)(r >> 7) * (Nn * Ll) + (long)node * Ll + (r & 127);
          if (c < 256)      o16a[token * Dd + c]         = (_Float16)v;
          else if (c < 512) o16b[token * Dd + (c - 256)] = (_Float16)v;
          else {
            int o = c - 512, hh = o >> 5, e = o & 31;
            long bnIdx = token >> 7;
            o16c[((bnIdx * Hh + hh) * Ee + e) * Ll + (token & 127)] = (_Float16)v;
          }
        }
      }
}

// ---------------- fused attention: QK^T -> softmax -> AV --------------------
// grid = B*N*H blocks, 256 threads (8 waves); wave w owns score rows 16w..16w+15.
__global__ __launch_bounds__(256) void attn_kernel(const _Float16* __restrict__ q16,
                                                   const _Float16* __restrict__ k16,
                                                   const _Float16* __restrict__ vt,
                                                   float* __restrict__ attn_out,
                                                   _Float16* __restrict__ ctx16) {
  __shared__ _Float16 k_s[128 * 40];     // [s][e]  (e=0..31, padded)
  __shared__ _Float16 v_s[32 * 136];     // [e][s]  (s=0..127, padded)
  __shared__ _Float16 a_s[128 * 136];    // normalized attn (f16) per wave strip
  const int tid = threadIdx.x, lane = tid & 31, wave = tid >> 5;
  const int bnh = blockIdx.x;            // 0..2047
  const int hh = bnh & (Hh - 1);
  const long bn = bnh >> 3;

  { // async-stage K tile (128x32 f16)
    int s = tid >> 1, half = tid & 1;
    const _Float16* src = k16 + ((long)bn * Ll + s) * Dd + hh * Ee + half * 16;
    unsigned dst = lds_addr32(k_s + s * 40 + half * 16);
    async_b128(dst, src); async_b128(dst + 16, src + 8);
  }
  { // async-stage V^T tile (32x128 f16)
    int e = tid >> 3, seg = tid & 7;
    const _Float16* src = vt + (((long)bn * Hh + hh) * Ee + e) * Ll + seg * 16;
    unsigned dst = lds_addr32(v_s + e * 136 + seg * 16);
    async_b128(dst, src); async_b128(dst + 16, src + 8);
  }
  // Q fragment straight from global (rows 16w..16w+15, K = E = 32)
  v16h aq;
  {
    long row = bn * Ll + wave * 16 + (lane & 15);
    int k0 = (lane < 16) ? 0 : 8;
    const _Float16* p = q16 + row * Dd + hh * Ee + k0;
    aq = mk16(*(const h8*)p, *(const h8*)(p + 16));
  }
  wait_async0();
  __syncthreads();

  // scores: 8 WMMAs -> wave holds its 16x128 strip in accumulators
  v8f c[8];
  const int rowl = lane & 15, kb0 = (lane < 16) ? 0 : 8;
#pragma unroll
  for (int t = 0; t < 8; ++t) {
#pragma unroll
    for (int i = 0; i < 8; ++i) c[t][i] = 0.f;
    const _Float16* p = k_s + (t * 16 + rowl) * 40 + kb0;
    c[t] = wmma16(aq, mk16(*(const h8*)p, *(const h8*)(p + 16)), c[t]);
  }

  // in-register softmax: row r = i + (lane>=16?8:0) lives in one half-wave
  const float scale = 0.17677669529663687f;  // 1/sqrt(32)
  float mx[8], sm[8];
#pragma unroll
  for (int i = 0; i < 8; ++i) {
    float m = -3.4e38f;
#pragma unroll
    for (int t = 0; t < 8; ++t) { c[t][i] *= scale; m = fmaxf(m, c[t][i]); }
    mx[i] = m;
  }
#pragma unroll
  for (int off = 1; off < 16; off <<= 1)
#pragma unroll
    for (int i = 0; i < 8; ++i) mx[i] = fmaxf(mx[i], __shfl_xor(mx[i], off, 32));
#pragma unroll
  for (int i = 0; i < 8; ++i) {
    float s = 0.f;
#pragma unroll
    for (int t = 0; t < 8; ++t) { c[t][i] = __expf(c[t][i] - mx[i]); s += c[t][i]; }
    sm[i] = s;
  }
#pragma unroll
  for (int off = 1; off < 16; off <<= 1)
#pragma unroll
    for (int i = 0; i < 8; ++i) sm[i] += __shfl_xor(sm[i], off, 32);
#pragma unroll
  for (int i = 0; i < 8; ++i) sm[i] = 1.f / sm[i];

  // write f32 attn to d_out; stage f16 attn in LDS for the AV GEMM
  const int rofs = (lane >= 16) ? 8 : 0;
  const long obase = (long)bnh * Ll * Ll;
#pragma unroll
  for (int t = 0; t < 8; ++t)
#pragma unroll
    for (int i = 0; i < 8; ++i) {
      float v = c[t][i] * sm[i];
      int rloc = wave * 16 + i + rofs, col = t * 16 + (lane & 15);
      attn_out[obase + (long)rloc * Ll + col] = v;
      a_s[rloc * 136 + col] = (_Float16)v;
    }
  __syncthreads();

  // AV: [16x128] x [32x128]^T -> 16x32 per wave (8 WMMAs)
  v8f cc[2];
#pragma unroll
  for (int nt = 0; nt < 2; ++nt)
#pragma unroll
    for (int i = 0; i < 8; ++i) cc[nt][i] = 0.f;
  const int ka0 = (lane < 16) ? 0 : 8;
#pragma unroll
  for (int kk = 0; kk < 128; kk += 32) {
    const _Float16* pa = a_s + (wave * 16 + rowl) * 136 + kk + ka0;
    v16h af = mk16(*(const h8*)pa, *(const h8*)(pa + 16));
#pragma unroll
    for (int nt = 0; nt < 2; ++nt) {
      const _Float16* pb = v_s + (nt * 16 + rowl) * 136 + kk + kb0;
      cc[nt] = wmma16(af, mk16(*(const h8*)pb, *(const h8*)(pb + 16)), cc[nt]);
    }
  }
#pragma unroll
  for (int nt = 0; nt < 2; ++nt)
#pragma unroll
    for (int i = 0; i < 8; ++i) {
      int l = wave * 16 + i + rofs;
      int e = nt * 16 + (lane & 15);
      ctx16[((long)bn * Ll + l) * Dd + hh * Ee + e] = (_Float16)cc[nt][i];
    }
}

// ---------------- residual + layernorm (wave per row) -----------------------
template <bool FINAL>
__global__ __launch_bounds__(256) void ln_kernel(float* __restrict__ h,
                                                 const float* __restrict__ t,
                                                 const float* __restrict__ g,
                                                 const float* __restrict__ b,
                                                 _Float16* __restrict__ hf,
                                                 float* __restrict__ outF) {
  const int lane = threadIdx.x & 31, wave = threadIdx.x >> 5;
  const long row = (long)blockIdx.x * 8 + wave;
  float v[8];
  float s = 0.f;
#pragma unroll
  for (int i = 0; i < 8; ++i) {
    long idx = row * Dd + i * 32 + lane;
    v[i] = h[idx];
    if (!FINAL) v[i] += t[idx];
    s += v[i];
  }
#pragma unroll
  for (int off = 16; off > 0; off >>= 1) s += __shfl_xor(s, off, 32);
  const float mu = s * (1.f / 256.f);
  float q = 0.f;
#pragma unroll
  for (int i = 0; i < 8; ++i) { float d = v[i] - mu; q += d * d; }
#pragma unroll
  for (int off = 16; off > 0; off >>= 1) q += __shfl_xor(q, off, 32);
  const float inv = rsqrtf(q * (1.f / 256.f) + 1e-5f);
#pragma unroll
  for (int i = 0; i < 8; ++i) {
    long idx = row * Dd + i * 32 + lane;
    float o = (v[i] - mu) * inv * g[i * 32 + lane] + b[i * 32 + lane];
    if (FINAL) outF[idx] = fmaxf(o, 0.f);
    else { h[idx] = o; hf[idx] = (_Float16)o; }
  }
}

// ---------------- launch ----------------------------------------------------

extern "C" void kernel_launch(void* const* d_in, const int* in_sizes, int n_in,
                              void* d_out, int out_size, void* d_ws, size_t ws_size,
                              hipStream_t stream) {
  (void)in_sizes; (void)n_in; (void)out_size; (void)ws_size;
  const float* x    = (const float*)d_in[0];
  const float* Wemb = (const float*)d_in[1];
  const float* bemb = (const float*)d_in[2];
  const float* pos  = (const float*)d_in[3];
  const float* Wq   = (const float*)d_in[4];
  const float* bq   = (const float*)d_in[5];
  const float* Wk   = (const float*)d_in[6];
  const float* bk   = (const float*)d_in[7];
  const float* Wv   = (const float*)d_in[8];
  const float* bv   = (const float*)d_in[9];
  const float* Wo   = (const float*)d_in[10];
  const float* bo   = (const float*)d_in[11];
  const float* W1   = (const float*)d_in[12];
  const float* W2   = (const float*)d_in[13];
  const float* ln1g = (const float*)d_in[14];
  const float* ln1b = (const float*)d_in[15];
  const float* ln2g = (const float*)d_in[16];
  const float* ln2b = (const float*)d_in[17];
  const float* lnfg = (const float*)d_in[18];
  const float* lnfb = (const float*)d_in[19];

  char* ws = (char*)d_ws;
  float*     h     = (float*)(ws + 0);                    // 33.5 MB
  _Float16*  hf    = (_Float16*)(ws + 33554432);          // 16.8 MB
  _Float16*  q16   = (_Float16*)(ws + 50331648);
  _Float16*  k16   = (_Float16*)(ws + 67108864);
  _Float16*  vt    = (_Float16*)(ws + 83886080);
  _Float16*  ctx16 = (_Float16*)(ws + 100663296);
  float*     tmp   = (float*)(ws + 117440512);            // 33.5 MB
  _Float16*  y16   = (_Float16*)(ws + 150994944);         // 67 MB
  _Float16*  Wqkv  = (_Float16*)(ws + 218103808);         // 25.2 MB
  _Float16*  Wo16  = (_Float16*)(ws + 243269632);
  _Float16*  W116  = (_Float16*)(ws + 243531776);
  _Float16*  W216  = (_Float16*)(ws + 244580352);
  float*     bqkv  = (float*)(ws + 245628928);

  float* outF  = (float*)d_out;
  float* attnF = outF + (long)NTOK * Dd;                  // attn maps after enc_out

  // --- weight prep ---
  prep_qkvw<<<16384, 256, 0, stream>>>(Wq, Wqkv, 0);
  prep_qkvw<<<16384, 256, 0, stream>>>(Wk, Wqkv, 1);
  prep_qkvw<<<16384, 256, 0, stream>>>(Wv, Wqkv, 2);
  prep_bias<<<64, 256, 0, stream>>>(bq, bqkv, 0);
  prep_bias<<<64, 256, 0, stream>>>(bk, bqkv, 1);
  prep_bias<<<64, 256, 0, stream>>>(bv, bqkv, 2);
  prep_cvt<<<512, 256, 0, stream>>>(Wo, Wo16, (long)NLAY * Dd * Dd);
  prep_cvt<<<2048, 256, 0, stream>>>(W1, W116, (long)NLAY * DFF * Dd);
  prep_cvt<<<2048, 256, 0, stream>>>(W2, W216, (long)NLAY * Dd * DFF);

  // --- embedding ---
  embed_kernel<<<NTOK, 256, 0, stream>>>(x, Wemb, bemb, pos, h, hf);

  for (int l = 0; l < NLAY; ++l) {
    // QKV: per node M=1024, N=768, K=256
    gemm_nt<EPI_QKV><<<dim3(8, 6, Nn), 256, 0, stream>>>(
        hf, Wqkv + (long)l * Nn * 768 * 256, bqkv + (long)l * Nn * 768,
        nullptr, q16, k16, vt, 1024, 768, 256);
    // fused attention
    attn_kernel<<<Bb * Nn * Hh, 256, 0, stream>>>(
        q16, k16, vt, attnF + (long)l * Bb * Nn * Hh * Ll * Ll, ctx16);
    // output projection: M=32768, N=256, K=256
    gemm_nt<EPI_F32><<<dim3(256, 2, 1), 256, 0, stream>>>(
        ctx16, Wo16 + (long)l * Dd * Dd, bo + (long)l * Dd,
        tmp, nullptr, nullptr, nullptr, NTOK, Dd, Dd);
    ln_kernel<false><<<NTOK / 8, 256, 0, stream>>>(h, tmp, ln1g + l * Dd, ln1b + l * Dd, hf, nullptr);
    // FFN1: relu, M=32768, N=1024, K=256
    gemm_nt<EPI_RELU_F16><<<dim3(256, 8, 1), 256, 0, stream>>>(
        hf, W116 + (long)l * DFF * Dd, nullptr,
        nullptr, y16, nullptr, nullptr, NTOK, DFF, Dd);
    // FFN2: M=32768, N=256, K=1024 (bias-free)
    gemm_nt<EPI_F32><<<dim3(256, 2, 1), 256, 0, stream>>>(
        y16, W216 + (long)l * Dd * DFF, nullptr,
        tmp, nullptr, nullptr, nullptr, NTOK, Dd, DFF);
    ln_kernel<false><<<NTOK / 8, 256, 0, stream>>>(h, tmp, ln2g + l * Dd, ln2b + l * Dd, hf, nullptr);
  }
  // final LN + relu -> enc_out
  ln_kernel<true><<<NTOK / 8, 256, 0, stream>>>(h, nullptr, lnfg, lnfb, nullptr, outF);
}
